// MixtureOfExperts_34703335752395
// MI455X (gfx1250) — matrix operand order repaired and checked by reading, not compiled, not run on hardware
//
#include <hip/hip_runtime.h>
#include <hip/hip_bf16.h>
#include <math.h>

typedef __bf16 bf16_t;
typedef __attribute__((ext_vector_type(16))) __bf16 v16bf;
typedef __attribute__((ext_vector_type(8)))  __bf16 v8bf;
typedef __attribute__((ext_vector_type(8)))  float  v8f;

#define NTOK 16384   // B*T
#define DDIM 512
#define NEXP 4
#define TOPK 2
#define HDIM 2048
#define BM   32      // tokens per MoE block
#define TILES_PER_E (NTOK / BM)   // 512

__device__ __forceinline__ float fast_tanh(float x) {
  float r;
  asm volatile("v_tanh_f32 %0, %1\n\tv_nop" : "=v"(r) : "v"(x));
  return r;
}

// ---------------- convert fp32 -> bf16 (x, w1, w2), packed dword stores ----------------
__global__ void cvt_kernel(const float* __restrict__ x, const float* __restrict__ w1,
                           const float* __restrict__ w2,
                           bf16_t* __restrict__ xb, bf16_t* __restrict__ w1b,
                           bf16_t* __restrict__ w2b) {
  int i = blockIdx.x * blockDim.x + threadIdx.x;
  int stride = gridDim.x * blockDim.x;
  const float2* x2  = (const float2*)x;
  const float2* w12 = (const float2*)w1;
  const float2* w22 = (const float2*)w2;
  uint32_t* xo  = (uint32_t*)xb;
  uint32_t* w1o = (uint32_t*)w1b;
  uint32_t* w2o = (uint32_t*)w2b;
  union Pk { bf16_t h[2]; uint32_t u; };
  for (int p = i; p < NTOK * DDIM / 2; p += stride) {
    float2 f = x2[p];
    Pk pk; pk.h[0] = (bf16_t)f.x; pk.h[1] = (bf16_t)f.y;
    xo[p] = pk.u;
  }
  for (int p = i; p < NEXP * HDIM * DDIM / 2; p += stride) {
    float2 a = w12[p], b = w22[p];
    Pk pa; pa.h[0] = (bf16_t)a.x; pa.h[1] = (bf16_t)a.y;
    Pk pb; pb.h[0] = (bf16_t)b.x; pb.h[1] = (bf16_t)b.y;
    w1o[p] = pa.u;
    w2o[p] = pb.u;
  }
}

// ---------------- routing: gate, top-2, softmax, expert lists ----------------
__global__ void route_kernel(const float* __restrict__ x, const float* __restrict__ gw,
                             int* __restrict__ cnt, int* __restrict__ entries,
                             float* __restrict__ wlist, float* __restrict__ gts) {
  int tok  = (blockIdx.x * blockDim.x + threadIdx.x) >> 5;  // one wave32 per token
  int lane = threadIdx.x & 31;
  if (tok >= NTOK) return;
  const float* xr = x + (long)tok * DDIM;
  float a0 = 0.f, a1 = 0.f, a2 = 0.f, a3 = 0.f;
  for (int j = lane; j < DDIM; j += 32) {
    float xv = xr[j];
    a0 += xv * gw[0 * DDIM + j];
    a1 += xv * gw[1 * DDIM + j];
    a2 += xv * gw[2 * DDIM + j];
    a3 += xv * gw[3 * DDIM + j];
  }
  for (int off = 16; off; off >>= 1) {
    a0 += __shfl_xor(a0, off, 32);
    a1 += __shfl_xor(a1, off, 32);
    a2 += __shfl_xor(a2, off, 32);
    a3 += __shfl_xor(a3, off, 32);
  }
  if (lane == 0) {
    float g[4] = {a0, a1, a2, a3};
    int i0 = 0;
    for (int e = 1; e < 4; ++e) if (g[e] > g[i0]) i0 = e;      // ties -> lowest index
    int i1 = -1;
    for (int e = 0; e < 4; ++e) if (e != i0 && (i1 < 0 || g[e] > g[i1])) i1 = e;
    float ex = expf(g[i1] - g[i0]);          // softmax over the 2 selected (max-sub)
    float inv = 1.0f / (1.0f + ex);
    float w0 = inv, w1v = ex * inv;
    int p0 = atomicAdd(cnt + i0, 1);
    entries[i0 * NTOK + p0] = tok * 2 + 0;   // slot 0
    wlist[i0 * NTOK + p0]   = w0;
    int p1 = atomicAdd(cnt + i1, 1);
    entries[i1 * NTOK + p1] = tok * 2 + 1;   // slot 1
    wlist[i1 * NTOK + p1]   = w1v;
    gts[tok] = a0 + a1 + a2 + a3;            // for balancing loss mean
  }
}

// ---------------- balancing loss (deterministic tree reduction) ----------------
__global__ void loss_kernel(const float* __restrict__ gts, float* __restrict__ out_loss) {
  __shared__ float sm[256];
  float s = 0.f;
  for (int i = threadIdx.x; i < NTOK; i += 256) s += gts[i];
  sm[threadIdx.x] = s;
  __syncthreads();
  for (int off = 128; off; off >>= 1) {
    if (threadIdx.x < off) sm[threadIdx.x] += sm[threadIdx.x + off];
    __syncthreads();
  }
  if (threadIdx.x == 0) {
    float m = sm[0] / (float)(NTOK * NEXP);
    *out_loss = m * logf(m + 0.1f);
  }
}

// ---------------- fused expert MLP: gather -> GEMM -> GELU -> GEMM -> scale ----------------
__launch_bounds__(256)
__global__ void moe_kernel(const bf16_t* __restrict__ xb,
                           const bf16_t* __restrict__ w1b,
                           const bf16_t* __restrict__ w2b,
                           const float* __restrict__ b1,
                           const float* __restrict__ b2,
                           const int* __restrict__ cnt,
                           const int* __restrict__ entries,
                           const float* __restrict__ wlist,
                           float* __restrict__ obuf) {
  const int e    = blockIdx.x / TILES_PER_E;
  const int tile = blockIdx.x % TILES_PER_E;
  const int ce   = cnt[e];
  const int base = tile * BM;
  if (base >= ce) return;                      // uniform across block
  int rows = ce - base;
  if (rows > BM) rows = BM;

  __shared__ __align__(16) bf16_t sX[BM][520];   // x tile, padded stride (bank spread)
  __shared__ __align__(16) bf16_t sH[2][BM][72]; // double-buffered gelu(h) chunk
  __shared__ int   sEnt[BM];
  __shared__ float sWgt[BM];

  const int tid = threadIdx.x;
  if (tid < BM) {
    if (tid < rows) {
      sEnt[tid] = entries[e * NTOK + base + tid];
      sWgt[tid] = wlist[e * NTOK + base + tid];
    } else {
      sEnt[tid] = 0;
      sWgt[tid] = 0.f;
    }
  }
  __syncthreads();

  // gather x rows (bf16) into LDS, 4 bytes at a time
  for (int p = tid; p < BM * (DDIM / 2); p += 256) {
    int r  = p >> 8;                 // DDIM/2 == 256 dwords per row
    int c2 = p & 255;
    int tok = sEnt[r] >> 1;
    const uint32_t* src = (const uint32_t*)(xb + (long)tok * DDIM);
    *(uint32_t*)(&sX[r][c2 * 2]) = src[c2];
  }
  __syncthreads();

  const int lane   = tid & 31;
  const int w      = tid >> 5;       // 8 waves
  const int laneHi = lane >> 4;      // 0/1 half-wave
  const int lane15 = lane & 15;

  // phase A: wave w owns h-tile (miA, hiA) of the [BM x 64] chunk
  const int miA = w & 1, hiA = w >> 1;
  // phase B: wave w owns out tiles (miB, cols qB*128 .. +127)
  const int miB = w & 1, qB = w >> 1;

  v8f accB[8];
  for (int t = 0; t < 8; ++t) accB[t] = {};

  const long w1eBase = (long)e * HDIM * DDIM;
  const long w2eBase = (long)e * DDIM * HDIM;

  for (int ch = 0; ch < HDIM / 64; ++ch) {
    const int buf = ch & 1;
    // ---- phase A: h = x @ w1[e]^T for 64 hidden cols; two independent chains ----
    v8f accA0 = {}, accA1 = {};
    const int hrow = ch * 64 + hiA * 16 + lane15;          // B column -> w1 row
    const bf16_t* w1row = w1b + w1eBase + (long)hrow * DDIM;
    if (ch + 1 < HDIM / 64)
      __builtin_prefetch(w1row + 64 * DDIM, 0, 1);         // next chunk's w1 rows
#pragma unroll 4
    for (int ks = 0; ks < DDIM / 32; ks += 2) {
      union { v16bf v; v8bf h[2]; } fa0, fb0, fa1, fb1;
      // A frag (16x32 bf16): lanes0-15 k 0..7 & 16..23, lanes16-31 k 8..15 & 24..31
      int ar = miA * 16 + lane15;
      int ak = ks * 32 + laneHi * 8;
      fa0.h[0] = *(const v8bf*)(&sX[ar][ak]);
      fa0.h[1] = *(const v8bf*)(&sX[ar][ak + 16]);
      fa1.h[0] = *(const v8bf*)(&sX[ar][ak + 32]);
      fa1.h[1] = *(const v8bf*)(&sX[ar][ak + 48]);
      // B frag (32x16 bf16): lanes0-15 k 0..7 & 16..23, lanes16-31 k 8..15 & 24..31 (col = lane15)
      int bk = ks * 32 + laneHi * 8;
      fb0.h[0] = *(const v8bf*)(w1row + bk);
      fb0.h[1] = *(const v8bf*)(w1row + bk + 16);
      fb1.h[0] = *(const v8bf*)(w1row + bk + 32);
      fb1.h[1] = *(const v8bf*)(w1row + bk + 48);
      accA0 = __builtin_amdgcn_wmma_f32_16x16x32_bf16(false, fa0.v, false, fb0.v,
                                                      (short)0, accA0, false, false);
      accA1 = __builtin_amdgcn_wmma_f32_16x16x32_bf16(false, fa1.v, false, fb1.v,
                                                      (short)0, accA1, false, false);
    }
    // bias + GPT-2 gelu (hardware v_tanh_f32), write bf16 h-chunk to LDS
    const float b1v = b1[e * HDIM + ch * 64 + hiA * 16 + lane15];
#pragma unroll
    for (int r = 0; r < 8; ++r) {
      float v = accA0[r] + accA1[r] + b1v;
      float inner = 0.7978845608028654f * fmaf(0.044715f * v * v, v, v);
      float g = 0.5f * v * (1.0f + fast_tanh(inner));
      int row = miA * 16 + r + laneHi * 8;
      sH[buf][row][hiA * 16 + lane15] = (bf16_t)g;
    }
    __syncthreads();   // phase-A writes of sH[buf] visible; prior-chunk reads done
    // ---- phase B: out += gelu(h) @ w2[e]^T ----
    for (int ks = 0; ks < 2; ++ks) {
      union { v16bf v; v8bf h[2]; } fa;
      int ar = miB * 16 + lane15;
      int ak = ks * 32 + laneHi * 8;
      fa.h[0] = *(const v8bf*)(&sH[buf][ar][ak]);
      fa.h[1] = *(const v8bf*)(&sH[buf][ar][ak + 16]);
      const int hk = ch * 64 + ks * 32 + laneHi * 8;
#pragma unroll
      for (int t = 0; t < 8; ++t) {
        int dcol = qB * 128 + t * 16 + lane15;             // B column -> w2 row
        const bf16_t* w2row = w2b + w2eBase + (long)dcol * HDIM + hk;
        union { v16bf v; v8bf h[2]; } fb;
        fb.h[0] = *(const v8bf*)(w2row);
        fb.h[1] = *(const v8bf*)(w2row + 16);
        accB[t] = __builtin_amdgcn_wmma_f32_16x16x32_bf16(false, fa.v, false, fb.v,
                                                          (short)0, accB[t], false, false);
      }
    }
  }

  // epilogue: (o + b2) * combine_weight -> per-(token,slot) buffer (deterministic)
  for (int t = 0; t < 8; ++t) {
    int dcol = qB * 128 + t * 16 + lane15;
    float b2v = b2[e * DDIM + dcol];
#pragma unroll
    for (int r = 0; r < 8; ++r) {
      int row = miB * 16 + r + laneHi * 8;
      if (row < rows) {
        int entry = sEnt[row];
        obuf[(long)entry * DDIM + dcol] = (accB[t][r] + b2v) * sWgt[row];
      }
    }
  }
}

// ---------------- combine the two expert slots per token ----------------
__global__ void combine_kernel(const float* __restrict__ obuf, float* __restrict__ out) {
  long i = (long)blockIdx.x * blockDim.x + threadIdx.x;
  const long total = (long)NTOK * DDIM;
  if (i < total) {
    long n = i / DDIM;
    long d = i - n * DDIM;
    long b = (2 * n) * DDIM + d;
    out[i] = obuf[b] + obuf[b + DDIM];
  }
}

extern "C" void kernel_launch(void* const* d_in, const int* in_sizes, int n_in,
                              void* d_out, int out_size, void* d_ws, size_t ws_size,
                              hipStream_t stream) {
  const float* x  = (const float*)d_in[0];
  const float* gw = (const float*)d_in[1];
  const float* w1 = (const float*)d_in[2];
  const float* b1 = (const float*)d_in[3];
  const float* w2 = (const float*)d_in[4];
  const float* b2 = (const float*)d_in[5];
  float* out = (float*)d_out;

  char* ws = (char*)d_ws;
  size_t off = 0;
  auto alloc = [&](size_t bytes) -> char* {
    char* p = ws + off;
    off += (bytes + 255) & ~(size_t)255;
    return p;
  };
  bf16_t* xb   = (bf16_t*)alloc((size_t)NTOK * DDIM * 2);          // 16.8 MB
  bf16_t* w1b  = (bf16_t*)alloc((size_t)NEXP * HDIM * DDIM * 2);   // 8.4 MB
  bf16_t* w2b  = (bf16_t*)alloc((size_t)NEXP * HDIM * DDIM * 2);   // 8.4 MB
  float*  obuf = (float*)alloc((size_t)NTOK * TOPK * DDIM * 4);    // 67 MB
  int*    ent  = (int*)alloc((size_t)NEXP * NTOK * 4);
  float*  wl   = (float*)alloc((size_t)NEXP * NTOK * 4);
  int*    cnt  = (int*)alloc(256);
  float*  gts  = (float*)alloc((size_t)NTOK * 4);

  hipMemsetAsync(cnt, 0, 256, stream);
  cvt_kernel<<<512, 256, 0, stream>>>(x, w1, w2, xb, w1b, w2b);
  route_kernel<<<NTOK / 8, 256, 0, stream>>>(x, gw, cnt, ent, wl, gts);
  loss_kernel<<<1, 256, 0, stream>>>(gts, out + (size_t)NTOK * DDIM);
  moe_kernel<<<NEXP * TILES_PER_E, 256, 0, stream>>>(xb, w1b, w2b, b1, b2, cnt, ent, wl, obuf);
  combine_kernel<<<((long)NTOK * DDIM + 255) / 256, 256, 0, stream>>>(obuf, out);
}
